// SimilarityLoss_12962211299655
// MI455X (gfx1250) — matrix operand order repaired and checked by reading, not compiled, not run on hardware
//
#include <hip/hip_runtime.h>
#include <math.h>

// ---------------- problem constants ----------------
constexpr float G1 = 4.0f, G2 = 5.0f, G3 = 10.0f;
constexpr int B_  = 32;
constexpr int M_  = 36;
constexpr int D_  = 512;
constexpr int T_  = 24;
constexpr int TP  = 32;      // T padded to 32 rows (rows 24..31 zero)
constexpr int P_  = 49;      // pixels per region
constexpr int N_  = 1764;    // M_*P_
constexpr int NP  = 1792;    // N_ padded to multiple of 64
constexpr int BLK = 64;      // columns per block iteration (4 waves x 16)
constexpr int NBLK = NP / BLK;
constexpr int NPAIR = 352;   // 32 diag + 160 dq + 160 qd

typedef _Float16 v16h __attribute__((ext_vector_type(16)));
typedef _Float16 v8h  __attribute__((ext_vector_type(8)));
typedef float    v8f  __attribute__((ext_vector_type(8)));

__device__ __forceinline__ v8f wmma_f16(v16h a, v16h b, v8f c) {
  // D = A(16x32 f16) * B(32x16 f16) + C(16x16 f32)
  return __builtin_amdgcn_wmma_f32_16x16x32_f16(
      /*neg_a=*/false, a, /*neg_b=*/false, b,
      /*c_mod=*/(short)0, c, /*reuse_a=*/false, /*reuse_b=*/false);
}

// build a 16-f16 fragment from two contiguous 16-byte chunks
__device__ __forceinline__ v16h ldfrag(const _Float16* p0, const _Float16* p1) {
  v16h r;
  reinterpret_cast<v8h*>(&r)[0] = *reinterpret_cast<const v8h*>(p0);
  reinterpret_cast<v8h*>(&r)[1] = *reinterpret_cast<const v8h*>(p1);
  return r;
}

// ---------------- pack text -> f16 (padded to 32 rows) + norms ----------------
__global__ __launch_bounds__(256)
void pack_text_kernel(const float* __restrict__ text,
                      _Float16* __restrict__ eh, float* __restrict__ enorm) {
  int bt = blockIdx.x;            // b*32 + t
  int b = bt >> 5, t = bt & 31;
  int tid = threadIdx.x;
  __shared__ float red[256];
  float s = 0.f;
  for (int d = tid; d < D_; d += 256) {
    float v = (t < T_) ? text[((size_t)b * T_ + t) * D_ + d] : 0.f;
    eh[(size_t)bt * D_ + d] = (_Float16)v;
    s += v * v;
  }
  red[tid] = s; __syncthreads();
  for (int st = 128; st > 0; st >>= 1) { if (tid < st) red[tid] += red[tid + st]; __syncthreads(); }
  if (tid == 0) enorm[bt] = (t < T_) ? sqrtf(red[0]) : 1.f;
}

// ------- pack image -> v_all (row major f16), v_allT (d-major f16), row norms -------
__global__ __launch_bounds__(256)
void pack_image_kernel(const float* __restrict__ image,
                       _Float16* __restrict__ vh, _Float16* __restrict__ vhT,
                       float* __restrict__ vnorm) {
  int bn = blockIdx.x;            // b*NP + n
  int b = bn / NP, n = bn % NP;
  int m = n / P_, pp = n % P_, hw = pp / 7, hr = pp % 7;
  bool valid = (n < N_);
  int tid = threadIdx.x;
  __shared__ float red[256];
  float s = 0.f;
  for (int d = tid; d < D_; d += 256) {
    // v_all[b,n,d] = image[b, m, d, hr, hw]  (transpose (0,1,4,3,2))
    float v = valid ? image[((((size_t)b * M_ + m) * D_ + d) * 7 + hr) * 7 + hw] : 0.f;
    vh[(size_t)bn * D_ + d] = (_Float16)v;
    vhT[((size_t)b * D_ + d) * NP + n] = (_Float16)v;
    s += v * v;
  }
  red[tid] = s; __syncthreads();
  for (int st = 128; st > 0; st >>= 1) { if (tid < st) red[tid] += red[tid + st]; __syncthreads(); }
  if (tid == 0) vnorm[bn] = valid ? sqrtf(red[0]) : 1.f;
}

// ---------------- LDS layout for match kernel (dynamic) ----------------
constexpr int OFF_W    = 0;                    // w tile:   [32][64] f16  (4096 B)
constexpr int OFF_EN   = 4096;                 // exp(nsm): [32][64] f32  (8192 B)
constexpr int OFF_AB   = 12288;                // A_beta:   [24][36] f32  (3456 B)
constexpr int OFF_P1   = 15744;                // [32] f32
constexpr int OFF_P2   = 15872;                // [32] f32
constexpr int OFF_EPN  = 16000;                // [36] f32
constexpr int OFF_REF  = 16144;                // [36] f32
constexpr int OFF_BETA = 16288;                // [24][36] f32 (3456 B)
constexpr int OFF_EPR  = 19744;                // e_prime: [36][512] f32 (73728 B)
constexpr int SMEM_MATCH = OFF_EPR + M_ * D_ * 4;   // 93472 B

// ---------------- one workgroup per (v_image, e_caption) pair ----------------
__global__ __launch_bounds__(128)
void match_kernel(const _Float16* __restrict__ vh, const _Float16* __restrict__ vhT,
                  const _Float16* __restrict__ eh, const float* __restrict__ vnorm,
                  const float* __restrict__ enorm, const int* __restrict__ spinds,
                  float* __restrict__ score, float* __restrict__ beta_out) {
  extern __shared__ char smem[];
  _Float16* w_lds  = (_Float16*)(smem + OFF_W);
  float*    en_lds = (float*)(smem + OFF_EN);
  float*    A_beta = (float*)(smem + OFF_AB);
  float*    P1     = (float*)(smem + OFF_P1);
  float*    P2     = (float*)(smem + OFF_P2);
  float*    epn2   = (float*)(smem + OFF_EPN);
  float*    refsum = (float*)(smem + OFF_REF);
  float*    beta_l = (float*)(smem + OFF_BETA);
  float*    eprime = (float*)(smem + OFF_EPR);

  const int tid  = threadIdx.x;
  const int wave = tid >> 5;
  const int lane = tid & 31;
  const int mrow = lane & 15;
  const int lh   = lane >> 4;       // 0: lanes 0-15, 1: lanes 16-31

  // decode pair slot
  const int p = blockIdx.x;
  int vi, ei;
  if (p < 32)        { vi = p; ei = p; }
  else if (p < 192)  { int b = (p - 32) / 5, j = (p - 32) % 5;  vi = b; ei = spinds[b * 5 + j]; }
  else               { int b = (p - 192) / 5, j = (p - 192) % 5; vi = spinds[b * 5 + j]; ei = b; }

  for (int i = tid; i < T_ * M_; i += 128) A_beta[i] = 0.f;
  if (tid < 32) { P1[tid] = 0.f; P2[tid] = 0.f; }
  if (tid < 36) { epn2[tid] = 0.f; refsum[tid] = 0.f; }

  const _Float16* eg  = eh  + (size_t)ei * TP * D_;
  const _Float16* vg  = vh  + (size_t)vi * NP * D_;
  const _Float16* vtg = vhT + (size_t)vi * D_ * NP;
  const int d0 = wave * 128;        // this wave's D-slice for vt accumulation

  v8f zero = {};
  v8f acc[2][8];                    // vt_acc tiles: rows(t) 0-15 / 16-31, 8x 16 d-cols
  #pragma unroll
  for (int r = 0; r < 2; ++r)
    #pragma unroll
    for (int j = 0; j < 8; ++j) acc[r][j] = zero;

  __syncthreads();

  for (int nb = 0; nb < NBLK; ++nb) {
    const int n0  = nb * BLK;
    const int nw0 = n0 + wave * 16;       // this wave's 16 columns
    if (nb + 1 < NBLK)
      __builtin_prefetch(vg + (size_t)(nw0 + BLK + mrow) * D_, 0, 1);

    // ---- GEMM 1: S tile (rows t=0..31, cols nw0..nw0+15), K = D ----
    v8f c0 = zero, c1 = zero;
    #pragma unroll 4
    for (int ks = 0; ks < D_ / 32; ++ks) {
      const int k0 = ks * 32;
      const _Float16* a0p = eg + (size_t)mrow * D_ + k0 + lh * 8;
      const _Float16* a1p = eg + (size_t)(16 + mrow) * D_ + k0 + lh * 8;
      v16h a0 = ldfrag(a0p, a0p + 16);           // A: 16x32 f16 layout
      v16h a1 = ldfrag(a1p, a1p + 16);
      const _Float16* bp = vg + (size_t)(nw0 + mrow) * D_ + k0 + lh * 8;
      v16h bf = ldfrag(bp, bp + 16);             // B: 32x16 (same K order as A)
      c0 = wmma_f16(a0, bf, c0);
      c1 = wmma_f16(a1, bf, c1);
    }

    // ---- column softmax over t (axis=0) for column n = nw0+mrow ----
    // lane holds: c0 -> t = {0..7 | 8..15}, c1 -> t = {16..23 | 24..31(zero rows)}
    float mx = -1e30f;
    #pragma unroll
    for (int i = 0; i < 8; ++i) mx = fmaxf(mx, c0[i]);
    if (lh == 0) {
      #pragma unroll
      for (int i = 0; i < 8; ++i) mx = fmaxf(mx, c1[i]);
    }
    mx = fmaxf(mx, __shfl_xor(mx, 16, 32));
    float e0[8], e1[8], ssum = 0.f;
    #pragma unroll
    for (int i = 0; i < 8; ++i) { e0[i] = expf(c0[i] - mx); ssum += e0[i]; }
    #pragma unroll
    for (int i = 0; i < 8; ++i) { e1[i] = (lh == 0) ? expf(c1[i] - mx) : 0.f; ssum += e1[i]; }
    ssum += __shfl_xor(ssum, 16, 32);
    const float inv = 1.f / ssum;
    const int ng  = nw0 + mrow;
    const bool valid = (ng < N_);
    const int col = wave * 16 + mrow;
    #pragma unroll
    for (int i = 0; i < 8; ++i) {
      float nsm0 = e0[i] * inv;                  // t = i + lh*8  (always a real row)
      int t0 = i + lh * 8;
      w_lds[t0 * BLK + col]  = (_Float16)(valid ? expf(G1 * nsm0) : 0.f);
      en_lds[t0 * BLK + col] = valid ? expf(nsm0) : 0.f;
      float nsm1 = e1[i] * inv;                  // t = 16 + i (+8 for padded rows)
      int t1 = 16 + i + lh * 8;
      bool v1 = valid && (lh == 0);
      w_lds[t1 * BLK + col]  = (_Float16)(v1 ? expf(G1 * nsm1) : 0.f);
      en_lds[t1 * BLK + col] = v1 ? expf(nsm1) : 0.f;
    }
    __syncthreads();

    // ---- beta accumulator: A[t][m] += exp(nsm) (row-owner, race-free) ----
    if (tid < T_) {
      #pragma unroll 8
      for (int c = 0; c < BLK; ++c) {
        int ngc = n0 + c;
        if (ngc < N_) A_beta[tid * M_ + ngc / P_] += en_lds[tid * BLK + c];
      }
    }

    // ---- GEMM 2: vt_acc += w(32 x 64blk) @ v(64blk x D), per-wave 128 d-cols ----
    #pragma unroll
    for (int kk = 0; kk < 2; ++kk) {
      const int kb = kk * 32;
      const _Float16* aw0p = w_lds + (size_t)mrow * BLK + kb + lh * 8;
      const _Float16* aw1p = w_lds + (size_t)(16 + mrow) * BLK + kb + lh * 8;
      v16h aw0 = ldfrag(aw0p, aw0p + 16);
      v16h aw1 = ldfrag(aw1p, aw1p + 16);
      #pragma unroll
      for (int j = 0; j < 8; ++j) {
        const int d = d0 + j * 16 + mrow;
        const _Float16* bp = vtg + (size_t)d * NP + n0 + kb + lh * 8;
        v16h bf = ldfrag(bp, bp + 16);           // same K order as A, via vhT layout
        acc[0][j] = wmma_f16(aw0, bf, acc[0][j]);
        acc[1][j] = wmma_f16(aw1, bf, acc[1][j]);
      }
    }
    __syncthreads();
  }

  // ---- R1 pieces: P1[t] = vt_acc[t].e[t], P2[t] = |vt_acc[t]|^2 (attn norm cancels) ----
  #pragma unroll
  for (int r = 0; r < 2; ++r) {
    #pragma unroll
    for (int i = 0; i < 8; ++i) {
      int t = i + lh * 8 + r * 16;
      float p1 = 0.f, p2 = 0.f;
      #pragma unroll
      for (int j = 0; j < 8; ++j) {
        float a  = acc[r][j][i];
        float ev = (float)eg[(size_t)t * D_ + d0 + j * 16 + mrow];
        p1 += a * ev; p2 += a * a;
      }
      if (t < T_) { atomicAdd(&P1[t], p1); atomicAdd(&P2[t], p2); }
    }
  }
  __syncthreads();

  // ---- beta[t][m] = A[t][m] / sum_t A[t][m] ----
  if (tid < M_) {
    float ls = 0.f;
    for (int t = 0; t < T_; ++t) ls += A_beta[t * M_ + tid];
    float invls = 1.f / ls;
    for (int t = 0; t < T_; ++t) {
      float bv = A_beta[t * M_ + tid] * invls;
      beta_l[t * M_ + tid] = bv;
      if (p < 32) beta_out[((size_t)p * T_ + t) * M_ + tid] = bv;
    }
  }
  __syncthreads();

  // ---- e_prime[m][d] = sum_t e[t][d] * beta[t][m]; epn2[m] = |e_prime[:,m]|^2 ----
  for (int idx = tid; idx < M_ * D_; idx += 128) {
    int m = idx >> 9, d = idx & (D_ - 1);
    float s = 0.f;
    for (int t = 0; t < T_; ++t) s += (float)eg[(size_t)t * D_ + d] * beta_l[t * M_ + m];
    eprime[(size_t)m * D_ + d] = s;
    atomicAdd(&epn2[m], s * s);
  }
  __syncthreads();

  // ---- refsum[m] = sum_p cos(v[n], e_prime[:,m]) ----
  for (int n = tid; n < N_; n += 128) {
    int m = n / P_;
    const _Float16* vr = vg + (size_t)n * D_;
    const float* epr = eprime + (size_t)m * D_;
    float s = 0.f;
    for (int d = 0; d < D_; ++d) s += (float)vr[d] * epr[d];
    float cosv = s / (vnorm[(size_t)vi * NP + n] * sqrtf(epn2[m]));
    atomicAdd(&refsum[m], cosv);
  }
  __syncthreads();

  if (tid == 0) {
    // R2 = logsumexp(G2 * mean_p cos) / G2
    float mx2 = -1e30f;
    for (int m = 0; m < M_; ++m) mx2 = fmaxf(mx2, G2 * refsum[m] / (float)P_);
    float s2 = 0.f;
    for (int m = 0; m < M_; ++m) s2 += expf(G2 * refsum[m] / (float)P_ - mx2);
    float R2 = (logf(s2) + mx2) / G2;
    // R1 = logsumexp(G2 * cos1) / G2
    float mx1 = -1e30f;
    for (int t = 0; t < T_; ++t) {
      float c = G2 * P1[t] / (sqrtf(P2[t]) * enorm[ei * TP + t]);
      mx1 = fmaxf(mx1, c);
    }
    float s1 = 0.f;
    for (int t = 0; t < T_; ++t) {
      float c = G2 * P1[t] / (sqrtf(P2[t]) * enorm[ei * TP + t]);
      s1 += expf(c - mx1);
    }
    float R1 = (logf(s1) + mx1) / G2;
    score[p] = R1 + R2;
  }
}

// ---------------- final reduction: losses + off-diagonal Gram regularizer ----------------
__global__ __launch_bounds__(256)
void reduce_kernel(const float* __restrict__ score, const float* __restrict__ beta,
                   float* __restrict__ out) {
  __shared__ float offsq[B_];
  int tid = threadIdx.x;
  if (tid < B_) offsq[tid] = 0.f;
  __syncthreads();
  for (int idx = tid; idx < B_ * T_ * T_; idx += 256) {
    int b = idx / (T_ * T_), r = idx % (T_ * T_), t = r / T_, s = r % T_;
    if (t == s) continue;
    const float* bt = beta + ((size_t)b * T_ + t) * M_;
    const float* bs = beta + ((size_t)b * T_ + s) * M_;
    float d = 0.f;
    for (int m = 0; m < M_; ++m) d += bt[m] * bs[m];
    atomicAdd(&offsq[b], d * d);
  }
  __syncthreads();
  if (tid == 0) {
    float loss_reg = 0.f;
    for (int b = 0; b < B_; ++b) loss_reg += sqrtf(offsq[b]);
    float loss1 = 0.f, loss2 = 0.f;
    for (int b = 0; b < B_; ++b) {
      float num = G3 * expf(score[b]);
      float pdq = 0.f, pqd = 0.f;
      for (int j = 0; j < 5; ++j) {
        pdq += G3 * expf(score[32 + b * 5 + j]);
        pqd += G3 * expf(score[192 + b * 5 + j]);
      }
      loss1 -= num / pdq;
      loss2 -= num / pqd;
    }
    out[0] = loss1 + loss2 + loss_reg;
  }
}

// ---------------- host launcher ----------------
extern "C" void kernel_launch(void* const* d_in, const int* in_sizes, int n_in,
                              void* d_out, int out_size, void* d_ws, size_t ws_size,
                              hipStream_t stream) {
  (void)in_sizes; (void)n_in; (void)out_size; (void)ws_size;
  const float* image = (const float*)d_in[0];
  const float* text  = (const float*)d_in[1];
  const int*   spind = (const int*)d_in[3];   // length_info (d_in[2]) == T, unused

  char* ws = (char*)d_ws;
  size_t off = 0;
  _Float16* vh   = (_Float16*)(ws + off); off += (size_t)B_ * NP * D_ * 2;   // 58.7 MB
  _Float16* vhT  = (_Float16*)(ws + off); off += (size_t)B_ * D_ * NP * 2;   // 58.7 MB
  _Float16* eh   = (_Float16*)(ws + off); off += (size_t)B_ * TP * D_ * 2;   // 1 MB
  float*    vnrm = (float*)(ws + off);    off += (size_t)B_ * NP * 4;
  float*    enrm = (float*)(ws + off);    off += (size_t)B_ * TP * 4;
  float*    scr  = (float*)(ws + off);    off += 2048;
  float*    betg = (float*)(ws + off);    off += (size_t)B_ * T_ * M_ * 4;

  pack_text_kernel<<<B_ * TP, 256, 0, stream>>>(text, eh, enrm);
  pack_image_kernel<<<B_ * NP, 256, 0, stream>>>(image, vh, vhT, vnrm);
  match_kernel<<<NPAIR, 128, SMEM_MATCH, stream>>>(vh, vhT, eh, vnrm, enrm, spind, scr, betg);
  reduce_kernel<<<1, 256, 0, stream>>>(scr, betg, (float*)d_out);
}
